// GCNEncoderWithGate_55027120996894
// MI455X (gfx1250) — matrix operand order repaired and checked by reading, not compiled, not run on hardware
//
#include <hip/hip_runtime.h>
#include <hip/hip_bf16.h>

typedef __attribute__((ext_vector_type(16))) __bf16 v16bf;
typedef __attribute__((ext_vector_type(8)))  __bf16 v8bf;
typedef __attribute__((ext_vector_type(8)))  float  v8f;

#define NNODES 100000
#define NEDGES 1600000
#define DMODEL 128
#define WT_PITCH 136   // LDS row pitch in bf16 elems: 272B -> 4-bank advance per row, conflict-free b128 reads

// ---------------- input / output path modes for the fused GEMM ----------------
enum { IN_F32 = 0, IN_F32_BIAS_RELU = 1, IN_BF16 = 2 };
enum { OUT_GATE_BF16 = 0, OUT_F32 = 1 };

// C[r, 0:128] = epilogue( A[r, 0:128] @ W[0:128, 0:128] )
// Block = 256 threads = 8 waves; each wave owns a 16-row stripe, loops 8 col tiles.
// A fragments (4 k-steps of 32) are loaded once per wave and reused for all col tiles.
template <int IN_MODE, int OUT_MODE>
__global__ __launch_bounds__(256) void gcn_gemm128(
    const void* __restrict__ Ain,       // [nrows,128] f32 or bf16 (per IN_MODE)
    const float* __restrict__ in_bias,  // [128]  (IN_F32_BIAS_RELU only)
    const float* __restrict__ W,        // [128,128] f32, K-major (W[k*128+n])
    const float* __restrict__ ep_bias,  // [128]  (OUT_GATE_BF16 only)
    const float* __restrict__ Xgate,    // [nrows,128] f32 (OUT_GATE_BF16 only)
    void* __restrict__ Cout,            // [nrows,128] bf16 or f32 (per OUT_MODE)
    int nrows)
{
    __shared__ __bf16 Wt[DMODEL * WT_PITCH];  // transposed weights: Wt[n*PITCH + k]

    const int tid  = threadIdx.x;
    const int lane = tid & 31;
    const int wave = tid >> 5;

    // ---- stage W into LDS, transposed + converted to bf16 ----
    for (int i = tid; i < DMODEL * DMODEL; i += 256) {
        int n = i >> 7, k = i & 127;
        Wt[n * WT_PITCH + k] = (__bf16)W[k * DMODEL + n];
    }
    __syncthreads();

    const int rowBase = blockIdx.x * 128 + wave * 16;
    const int rA      = rowBase + (lane & 15);
    const int rAc     = rA < nrows ? rA : nrows - 1;     // clamp loads, guard stores
    const int koff    = (lane >> 4) * 8;                 // lane-half K offset in A layout

    // ---- load the wave's A fragments: 4 k-steps x v16bf, per WMMA bf16 A layout ----
    // lane<16 : elems 0..7 = K k0..k0+7, elems 8..15 = K k0+16..k0+23 (k0 = ks*32)
    // lane>=16: same with +8
    v16bf afrag[4];
#pragma unroll
    for (int ks = 0; ks < 4; ++ks) {
        if constexpr (IN_MODE == IN_BF16) {
            const __bf16* A = (const __bf16*)Ain + (size_t)rAc * DMODEL;
            v8bf lo = *(const v8bf*)(A + ks * 32 + koff);
            v8bf hi = *(const v8bf*)(A + ks * 32 + koff + 16);
            afrag[ks] = __builtin_shufflevector(lo, hi, 0, 1, 2, 3, 4, 5, 6, 7,
                                                8, 9, 10, 11, 12, 13, 14, 15);
        } else {
            const float* A = (const float*)Ain + (size_t)rAc * DMODEL;
            const int k0 = ks * 32 + koff;
            float t[16];
#pragma unroll
            for (int seg = 0; seg < 2; ++seg) {
                float4 u = *(const float4*)(A + k0 + seg * 16);
                float4 w = *(const float4*)(A + k0 + seg * 16 + 4);
                t[seg * 8 + 0] = u.x; t[seg * 8 + 1] = u.y;
                t[seg * 8 + 2] = u.z; t[seg * 8 + 3] = u.w;
                t[seg * 8 + 4] = w.x; t[seg * 8 + 5] = w.y;
                t[seg * 8 + 6] = w.z; t[seg * 8 + 7] = w.w;
            }
#pragma unroll
            for (int j = 0; j < 16; ++j) {
                int K = k0 + j + (j >= 8 ? 8 : 0);
                float v = t[j];
                if constexpr (IN_MODE == IN_F32_BIAS_RELU)
                    v = fmaxf(v + in_bias[K], 0.0f);
                afrag[ks][j] = (__bf16)v;
            }
        }
    }

    // ---- B fragment source position in LDS (per WMMA bf16 B layout) ----
    // lane<16 : column N = lane,    elems 0..7 = K k0..k0+7, elems 8..15 = K k0+16..k0+23
    // lane>=16: column N = lane-16, same with +8
    const int nB = lane & 15;
    const int kB = (lane >> 4) * 8;

#pragma unroll
    for (int c = 0; c < 8; ++c) {
        v8f acc = {};
#pragma unroll
        for (int ks = 0; ks < 4; ++ks) {
            const __bf16* bp = &Wt[(c * 16 + nB) * WT_PITCH + ks * 32 + kB];
            v8bf blo = *(const v8bf*)(bp);
            v8bf bhi = *(const v8bf*)(bp + 16);
            v16bf b = __builtin_shufflevector(blo, bhi, 0, 1, 2, 3, 4, 5, 6, 7,
                                              8, 9, 10, 11, 12, 13, 14, 15);
            acc = __builtin_amdgcn_wmma_f32_16x16x32_bf16(
                false, afrag[ks], false, b, (short)0, acc, false, false);
        }

        // ---- epilogue: C layout: elem j -> row rowBase + j + (lane>=16 ? 8:0), col c*16 + (lane&15)
        const int col  = c * 16 + (lane & 15);
        const int rEp0 = rowBase + ((lane >> 4) << 3);
#pragma unroll
        for (int j = 0; j < 8; ++j) {
            const int rr = rEp0 + j;
            if (rr < nrows) {
                if constexpr (OUT_MODE == OUT_GATE_BF16) {
                    float z = acc[j] + ep_bias[col];
                    float g = 1.0f / (1.0f + __expf(-z));
                    float v = Xgate[(size_t)rr * DMODEL + col] * g;
                    ((__bf16*)Cout)[(size_t)rr * DMODEL + col] = (__bf16)v;
                } else {
                    ((float*)Cout)[(size_t)rr * DMODEL + col] = acc[j];
                }
            }
        }
    }
}

// ---------------- graph-side kernels ----------------

__global__ void gcn_fill_deg(unsigned* deg, int n) {
    int i = blockIdx.x * blockDim.x + threadIdx.x;
    if (i < n) deg[i] = 1u;  // self loop
}

__global__ void gcn_count_deg(const int* __restrict__ dst, unsigned* deg, int e) {
    int i = blockIdx.x * blockDim.x + threadIdx.x;
    if (i < e) atomicAdd(&deg[dst[i]], 1u);
}

__global__ void gcn_make_dinv(const unsigned* __restrict__ deg, float* dinv, int n) {
    int i = blockIdx.x * blockDim.x + threadIdx.x;
    if (i < n) dinv[i] = rsqrtf((float)deg[i]);
}

// agg[i,:] = hw[i,:] * dinv[i]^2   (the self-loop term; also initializes agg)
__global__ void gcn_self_init(const float* __restrict__ hw, const float* __restrict__ dinv,
                              float* __restrict__ agg, int n) {
    int t = blockIdx.x * blockDim.x + threadIdx.x;
    if (t < n * 32) {
        int node = t >> 5;
        int c    = (t & 31) * 4;
        float w  = dinv[node];
        w *= w;
        float4 v = *(const float4*)(hw + (size_t)node * DMODEL + c);
        float4 o = {v.x * w, v.y * w, v.z * w, v.w * w};
        *(float4*)(agg + (size_t)node * DMODEL + c) = o;
    }
}

// one wave per edge: agg[d,:] += hw[s,:] * dinv[s] * dinv[d]
__global__ __launch_bounds__(256) void gcn_edge_scatter(
    const float* __restrict__ hw, const float* __restrict__ dinv,
    const int* __restrict__ src, const int* __restrict__ dst,
    float* __restrict__ agg, int e)
{
    int eid  = blockIdx.x * 8 + (threadIdx.x >> 5);
    int lane = threadIdx.x & 31;
    if (eid < e) {
        int s = src[eid], d = dst[eid];
        float nrm = dinv[s] * dinv[d];
        float4 v  = *(const float4*)(hw + (size_t)s * DMODEL + lane * 4);
        float* p  = agg + (size_t)d * DMODEL + lane * 4;
        unsafeAtomicAdd(p + 0, v.x * nrm);
        unsafeAtomicAdd(p + 1, v.y * nrm);
        unsafeAtomicAdd(p + 2, v.z * nrm);
        unsafeAtomicAdd(p + 3, v.w * nrm);
    }
}

__global__ void gcn_bias_inplace(float* __restrict__ out, const float* __restrict__ b, int total) {
    int i = blockIdx.x * blockDim.x + threadIdx.x;
    if (i < total) out[i] += b[i & (DMODEL - 1)];
}

// ---------------- launcher ----------------

extern "C" void kernel_launch(void* const* d_in, const int* in_sizes, int n_in,
                              void* d_out, int out_size, void* d_ws, size_t ws_size,
                              hipStream_t stream) {
    const float* x      = (const float*)d_in[0];             // [N,128]
    const int*   eidx   = (const int*)d_in[1];               // [2,E]
    const float* gate_W = (const float*)d_in[2];             // [128,128]
    const float* gate_b = (const float*)d_in[3];             // [128]
    const float* W1     = (const float*)d_in[4];
    const float* b1     = (const float*)d_in[5];
    const float* W2     = (const float*)d_in[6];
    const float* b2     = (const float*)d_in[7];
    const int* src = eidx;
    const int* dst = eidx + NEDGES;

    // workspace carving (256B aligned)
    char* ws = (char*)d_ws;
    auto align_up = [](size_t v) { return (v + 255) & ~(size_t)255; };
    unsigned* deg = (unsigned*)ws;            ws += align_up((size_t)NNODES * 4);
    float* dinv   = (float*)ws;               ws += align_up((size_t)NNODES * 4);
    __bf16* h0b   = (__bf16*)ws;              ws += align_up((size_t)NNODES * DMODEL * 2);
    float* hw     = (float*)ws;               ws += align_up((size_t)NNODES * DMODEL * 4);
    float* agg1   = (float*)ws;               ws += align_up((size_t)NNODES * DMODEL * 4);
    float* out    = (float*)d_out;

    const int gemmBlocks = (NNODES + 127) / 128;

    // degrees / normalization
    gcn_fill_deg<<<(NNODES + 255) / 256, 256, 0, stream>>>(deg, NNODES);
    gcn_count_deg<<<(NEDGES + 255) / 256, 256, 0, stream>>>(dst, deg, NEDGES);
    gcn_make_dinv<<<(NNODES + 255) / 256, 256, 0, stream>>>(deg, dinv, NNODES);

    // h0 = bf16( x * sigmoid(x @ gate_W + gate_b) )
    gcn_gemm128<IN_F32, OUT_GATE_BF16><<<gemmBlocks, 256, 0, stream>>>(
        x, nullptr, gate_W, gate_b, x, h0b, NNODES);

    // layer 1: hw1 = h0 @ W1
    gcn_gemm128<IN_BF16, OUT_F32><<<gemmBlocks, 256, 0, stream>>>(
        h0b, nullptr, W1, nullptr, nullptr, hw, NNODES);
    gcn_self_init<<<(NNODES * 32 + 255) / 256, 256, 0, stream>>>(hw, dinv, agg1, NNODES);
    gcn_edge_scatter<<<(NEDGES + 7) / 8, 256, 0, stream>>>(hw, dinv, src, dst, agg1, NEDGES);

    // layer 2: hw2 = relu(agg1 + b1) @ W2   (bias+relu fused into A path)
    gcn_gemm128<IN_F32_BIAS_RELU, OUT_F32><<<gemmBlocks, 256, 0, stream>>>(
        agg1, b1, W2, nullptr, nullptr, hw, NNODES);
    gcn_self_init<<<(NNODES * 32 + 255) / 256, 256, 0, stream>>>(hw, dinv, out, NNODES);
    gcn_edge_scatter<<<(NEDGES + 7) / 8, 256, 0, stream>>>(hw, dinv, src, dst, out, NEDGES);

    // out += b2
    gcn_bias_inplace<<<(NNODES * DMODEL + 255) / 256, 256, 0, stream>>>(out, b2, NNODES * DMODEL);
}
